// Block_3539053052091
// MI455X (gfx1250) — matrix operand order repaired and checked by reading, not compiled, not run on hardware
//
#include <hip/hip_runtime.h>

typedef __bf16 bf16_t;
typedef __attribute__((ext_vector_type(8)))  __bf16 v8bf;
typedef __attribute__((ext_vector_type(16))) __bf16 v16bf;
typedef __attribute__((ext_vector_type(8)))  float  v8f;

#define LN_EPS   1e-5f
#define ATTN_EPS 1e-5f

// ---------------------------------------------------------------- helpers ---

__device__ __forceinline__ v8f wmma_bf16(v16bf a, v16bf b, v8f c) {
  // D = A(16x32 bf16) * B(32x16 bf16) + C(16x16 f32)
  return __builtin_amdgcn_wmma_f32_16x16x32_bf16(
      /*neg_a=*/false, a, /*neg_b=*/false, b,
      /*c_mod=*/(short)0, c, /*reuse_a=*/false, /*reuse_b=*/false);
}

__device__ __forceinline__ v16bf cat8(v8bf lo, v8bf hi) {
  return __builtin_shufflevector(lo, hi, 0,1,2,3,4,5,6,7,8,9,10,11,12,13,14,15);
}

__device__ __forceinline__ float wave_rsum(float v) {
#pragma unroll
  for (int m = 16; m >= 1; m >>= 1) v += __shfl_xor(v, m, 32);
  return v;
}
__device__ __forceinline__ float wave_rmax(float v) {
#pragma unroll
  for (int m = 16; m >= 1; m >>= 1) v = fmaxf(v, __shfl_xor(v, m, 32));
  return v;
}

// ------------------------------------------- weight transpose + bf16 cast --
// src [K,N] f32 row-major  ->  dst [N,K] bf16 row-major (coalesced both ways)

__global__ void transpose_w_kernel(const float* __restrict__ src,
                                   bf16_t* __restrict__ dst, int K, int N) {
  __shared__ float tile[32][33];
  int nt = blockIdx.x * 32, kt = blockIdx.y * 32;
  int tx = threadIdx.x & 31, ty = threadIdx.x >> 5;   // 32 x 8
#pragma unroll
  for (int r = ty; r < 32; r += 8)
    tile[r][tx] = src[(size_t)(kt + r) * N + nt + tx];
  __syncthreads();
#pragma unroll
  for (int r = ty; r < 32; r += 8)
    dst[(size_t)(nt + r) * K + kt + tx] = (bf16_t)tile[tx][r];
}

// ------------------------------------ V head-transpose straight from vp ----
// vp [b, l, 2*DE] f32 (v = first DE)  ->  vT [b, h, DH, L] bf16

__global__ void transpose_v_kernel(const float* __restrict__ vp,
                                   bf16_t* __restrict__ vT,
                                   int L, int H, int DH, int DE2) {
  __shared__ float tile[32][33];
  int bh = blockIdx.z;
  int b = bh / H, h = bh % H;
  int d0 = blockIdx.x * 32, l0 = blockIdx.y * 32;
  int tx = threadIdx.x & 31, ty = threadIdx.x >> 5;
#pragma unroll
  for (int r = ty; r < 32; r += 8)
    tile[r][tx] = vp[(size_t)(b * L + l0 + r) * DE2 + (size_t)h * DH + d0 + tx];
  __syncthreads();
#pragma unroll
  for (int r = ty; r < 32; r += 8)
    vT[(((size_t)bh * DH) + d0 + r) * L + l0 + tx] = (bf16_t)tile[tx][r];
}

// ------------------------------------------------- 3x LayerNorm (shared mu) --

__global__ void ln3_kernel(const float* __restrict__ x,
                           const float* __restrict__ w0, const float* __restrict__ b0,
                           const float* __restrict__ w1, const float* __restrict__ b1,
                           const float* __restrict__ w2, const float* __restrict__ b2,
                           bf16_t* __restrict__ a0, bf16_t* __restrict__ a1,
                           bf16_t* __restrict__ a2, int D) {
  int row = blockIdx.x;
  int tid = threadIdx.x;
  const float* xr = x + (size_t)row * D;
  float v[4];
  float s = 0.f, ss = 0.f;
#pragma unroll
  for (int j = 0; j < 4; ++j) {
    int d = tid + 256 * j;
    v[j] = xr[d];
    s += v[j]; ss += v[j] * v[j];
  }
  s = wave_rsum(s); ss = wave_rsum(ss);
  __shared__ float red[2][8];
  int wid = tid >> 5, lane = tid & 31;
  if (lane == 0) { red[0][wid] = s; red[1][wid] = ss; }
  __syncthreads();
  s = 0.f; ss = 0.f;
#pragma unroll
  for (int j = 0; j < 8; ++j) { s += red[0][j]; ss += red[1][j]; }
  float mu = s / (float)D;
  float var = ss / (float)D - mu * mu;
  float rs = rsqrtf(var + LN_EPS);
#pragma unroll
  for (int j = 0; j < 4; ++j) {
    int d = tid + 256 * j;
    float xh = (v[j] - mu) * rs;
    size_t o = (size_t)row * D + d;
    a0[o] = (bf16_t)(xh * w0[d] + b0[d]);
    a1[o] = (bf16_t)(xh * w1[d] + b1[d]);
    a2[o] = (bf16_t)(xh * w2[d] + b2[d]);
  }
}

// --------------------------------------------------------------- bf16 GEMM --
// C[M,N] = A[M,K](bf16,row-major) @ Bt[N,K](bf16, pre-transposed) + bias.
// One wave computes a 16x64 tile; 2-stage software pipeline over K (K%64==0).
// Unsigned 32-bit offsets off uniform bases -> SADDR+voffset addressing.

__global__ void gemm_bf16_kernel(const bf16_t* __restrict__ A,
                                 const bf16_t* __restrict__ Bt,
                                 const float* __restrict__ bias,
                                 float* __restrict__ Cf,
                                 bf16_t* __restrict__ Cb,
                                 int M, int N, int K) {
  int wid  = threadIdx.x >> 5;
  int lane = threadIdx.x & 31;
  int tile = blockIdx.x * (blockDim.x >> 5) + wid;
  int ntiles = N >> 6;
  int mt = tile / ntiles, nt = tile % ntiles;
  if (mt >= (M >> 4)) return;
  unsigned m0 = (unsigned)mt << 4, n0 = (unsigned)nt << 6;
  unsigned half = (unsigned)lane >> 4, nloc = (unsigned)lane & 15u;
  unsigned uK = (unsigned)K;

  unsigned aoff  = (m0 + nloc) * uK + half * 8u;   // A layout: lanes hold M=lane&15
  unsigned boff0 = (n0 + nloc) * uK + half * 8u;  // + t*16*K per column group
  auto loadA = [&](unsigned k0) {
    const bf16_t* ap = A + (aoff + k0);
    return cat8(*(const v8bf*)ap, *(const v8bf*)(ap + 16));
  };
  auto loadB = [&](unsigned t, unsigned k0) {
    const bf16_t* bp = Bt + (boff0 + t * 16u * uK + k0);
    return cat8(*(const v8bf*)bp, *(const v8bf*)(bp + 16));
  };

  v8f acc[4] = {};
  v16bf a0 = loadA(0), a1;
  v16bf b0f[4], b1f[4];
#pragma unroll
  for (unsigned t = 0; t < 4; ++t) b0f[t] = loadB(t, 0);

  for (unsigned k0 = 0; k0 < uK; k0 += 64) {
    // issue stage-1 loads (k0+32), then consume stage-0
    a1 = loadA(k0 + 32);
#pragma unroll
    for (unsigned t = 0; t < 4; ++t) b1f[t] = loadB(t, k0 + 32);
#pragma unroll
    for (int t = 0; t < 4; ++t) acc[t] = wmma_bf16(a0, b0f[t], acc[t]);
    // issue stage-0 loads for next k-block (if any), then consume stage-1
    unsigned k2 = k0 + 64;
    if (k2 < uK) {
      a0 = loadA(k2);
#pragma unroll
      for (unsigned t = 0; t < 4; ++t) b0f[t] = loadB(t, k2);
    }
#pragma unroll
    for (int t = 0; t < 4; ++t) acc[t] = wmma_bf16(a1, b1f[t], acc[t]);
  }

#pragma unroll
  for (unsigned t = 0; t < 4; ++t) {
    unsigned col = n0 + t * 16u + nloc;
    float bv = bias ? bias[col] : 0.0f;
#pragma unroll
    for (unsigned r = 0; r < 8; ++r) {
      unsigned row = m0 + half * 8u + r;   // C layout: lanes16-31 hold M=8..15
      float val = acc[t][r] + bv;
      unsigned co = row * (unsigned)N + col;
      if (Cf) Cf[co] = val;
      if (Cb) Cb[co] = (bf16_t)val;
    }
  }
}

// ------------------------------------ softmax feature maps + smear + layout --

__global__ void qk_feat_kernel(const bf16_t* __restrict__ qg,
                               const bf16_t* __restrict__ kg,
                               const float* __restrict__ smear_factor,
                               const float* __restrict__ log_scale,
                               bf16_t* __restrict__ qh,
                               bf16_t* __restrict__ kh,
                               int Bn, int L, int H, int DH) {
  int w = blockIdx.x * (blockDim.x >> 5) + (threadIdx.x >> 5);
  int lane = threadIdx.x & 31;
  int h = w % H;
  int l = (w / H) % L;
  int b = w / (H * L);
  if (b >= Bn) return;
  int DE = H * DH;
  size_t rowg = ((size_t)(b * L + l)) * DE + (size_t)h * DH;   // [b*l, d_exp]
  size_t rowh = (((size_t)(b * H + h)) * L + l) * DH;          // [b,h,l,dh]
  float sinv = 1.0f / expf(log_scale[h]);
  float sm = 1.0f / (1.0f + expf(-smear_factor[h]));

  // q softmax
  float qv[4];
  float mx = -1e30f;
#pragma unroll
  for (int j = 0; j < 4; ++j) { qv[j] = (float)qg[rowg + j * 32 + lane]; mx = fmaxf(mx, qv[j]); }
  mx = wave_rmax(mx);
  float sum = 0.f;
#pragma unroll
  for (int j = 0; j < 4; ++j) { qv[j] = __expf(qv[j] - mx); sum += qv[j]; }
  sum = wave_rsum(sum);
  float rq = sinv / sum;
#pragma unroll
  for (int j = 0; j < 4; ++j) qh[rowh + j * 32 + lane] = (bf16_t)(qv[j] * rq);

  // k softmax (current)
  float kc[4];
  mx = -1e30f;
#pragma unroll
  for (int j = 0; j < 4; ++j) { kc[j] = (float)kg[rowg + j * 32 + lane]; mx = fmaxf(mx, kc[j]); }
  mx = wave_rmax(mx);
  sum = 0.f;
#pragma unroll
  for (int j = 0; j < 4; ++j) { kc[j] = __expf(kc[j] - mx); sum += kc[j]; }
  sum = wave_rsum(sum);
  float rc = 1.0f / sum;

  // k softmax (previous position), zero-padded at l==0
  float kp[4] = {0.f, 0.f, 0.f, 0.f};
  if (l > 0) {
    float mp = -1e30f;
#pragma unroll
    for (int j = 0; j < 4; ++j) { kp[j] = (float)kg[rowg - DE + j * 32 + lane]; mp = fmaxf(mp, kp[j]); }
    mp = wave_rmax(mp);
    float sp = 0.f;
#pragma unroll
    for (int j = 0; j < 4; ++j) { kp[j] = __expf(kp[j] - mp); sp += kp[j]; }
    sp = wave_rsum(sp);
    float rp = 1.0f / sp;
#pragma unroll
    for (int j = 0; j < 4; ++j) kp[j] *= rp;
  }
#pragma unroll
  for (int j = 0; j < 4; ++j) {
    float kv = ((1.0f - sm) * kc[j] * rc + sm * kp[j]) * sinv;
    kh[rowh + j * 32 + lane] = (bf16_t)kv;
  }
}

// ------------------------------------------------- flash-style causal attn --
// wave per (b,h, 16-query tile). Pipelined: S(p+1) (K loads + QK^T WMMAs +
// mask + LDS store) overlaps the O/rowsum WMMAs of step p via a 2-deep
// LDS S-tile ring buffer. Uniform per-(b,h) bases + unsigned 32-bit offsets.

__global__ void attn_kernel(const bf16_t* __restrict__ qh,
                            const bf16_t* __restrict__ kh,
                            const bf16_t* __restrict__ vT,
                            float* __restrict__ oh,
                            int L, int H) {
  __shared__ __align__(16) bf16_t sS[8][2][16][32];
  int wid  = threadIdx.x >> 5;
  int lane = threadIdx.x & 31;
  int gw = blockIdx.x * (blockDim.x >> 5) + wid;
  int nqt = L >> 4;
  int qt = gw % nqt;
  int bh = gw / nqt;
  int h  = bh % H;
  unsigned half = (unsigned)lane >> 4, n = (unsigned)lane & 15u;
  unsigned uL = (unsigned)L;

  bool spatial = (h < (H >> 1));
  float slope = spatial ? exp2f(-8.0f * (float)h / (float)((H >> 1) - 1)) : 0.0f;

  // wave-uniform bases (SGPR) + unsigned 32-bit lane offsets (VGPR)
  const bf16_t* qbh = qh + (size_t)bh * L * 128;
  const bf16_t* kbh = kh + (size_t)bh * L * 128;
  const bf16_t* vbh = vT + (size_t)bh * 128 * L;
  float*        obh = oh + (size_t)bh * L * 128;

  // Q fragments (16 queries x 128 dh) -> 4 A-frags of 16x32
  unsigned qoff = ((unsigned)qt * 16u + n) * 128u + half * 8u;
  v16bf qa[4];
#pragma unroll
  for (unsigned kk = 0; kk < 4; ++kk) {
    const bf16_t* rp = qbh + (qoff + kk * 32u);
    qa[kk] = cat8(*(const v8bf*)rp, *(const v8bf*)(rp + 16));
  }
  v16bf onesb;
#pragma unroll
  for (int e = 0; e < 16; ++e) onesb[e] = (bf16_t)1.0f;

  // compute S tile pair p (32 keys), mask, store bf16 into LDS parity `par`
  auto computeS = [&](int p, int par) {
#pragma unroll
    for (int t = 0; t < 2; ++t) {
      int kt = 2 * p + t;
      v8f s = {};
      if (kt <= qt) {
        unsigned koff = ((unsigned)kt * 16u + n) * 128u + half * 8u;
#pragma unroll
        for (unsigned kk = 0; kk < 4; ++kk) {
          const bf16_t* kp_ = kbh + (koff + kk * 32u);
          v16bf kb = cat8(*(const v8bf*)kp_, *(const v8bf*)(kp_ + 16));
          s = wmma_bf16(qa[kk], kb, s);
        }
      }
      int j = kt * 16 + (int)n;
#pragma unroll
      for (int r = 0; r < 8; ++r) {
        int i = qt * 16 + (int)half * 8 + r;
        float val = (kt <= qt) ? s[r] : 0.0f;
        if (j > i) val = 0.0f;
        else if (spatial) val *= __expf(-slope * (float)(i - j));
        sS[wid][par][half * 8 + r][t * 16 + n] = (bf16_t)val;
      }
    }
  };

  v8f oacc[8] = {};
  v8f racc = {};
  int P = qt >> 1;                          // last pair index
  unsigned voff0 = n * uL + half * 8u;     // + cc*16*L + p*32

  computeS(0, 0);
  for (int p = 0; p <= P; ++p) {
    int par = p & 1;
    // S(p) as A-fragment from LDS
    v16bf sa = cat8(*(const v8bf*)&sS[wid][par][n][half * 8],
                    *(const v8bf*)&sS[wid][par][n][16 + half * 8]);
    // issue all V fragment loads for step p
    v16bf vbf[8];
#pragma unroll
    for (unsigned cc = 0; cc < 8; ++cc) {
      const bf16_t* vp_ = vbh + (voff0 + cc * 16u * uL + (unsigned)p * 32u);
      vbf[cc] = cat8(*(const v8bf*)vp_, *(const v8bf*)(vp_ + 16));
    }
    // overlap: compute next S tile pair into the other parity buffer
    if (p < P) computeS(p + 1, par ^ 1);
    // O += S @ V ; rowsum += S @ ones
#pragma unroll
    for (int cc = 0; cc < 8; ++cc) oacc[cc] = wmma_bf16(sa, vbf[cc], oacc[cc]);
    racc = wmma_bf16(sa, onesb, racc);
  }

  // normalize by rowsum and store f32 [b,h,l,dh]
#pragma unroll
  for (unsigned r = 0; r < 8; ++r) {
    float inv = 1.0f / (racc[r] + ATTN_EPS);
    unsigned row = (unsigned)qt * 16u + half * 8u + r;
#pragma unroll
    for (unsigned cc = 0; cc < 8; ++cc)
      obh[row * 128u + cc * 16u + n] = oacc[cc][r] * inv;
  }
}

// -------------------------------------------------------------- SiLU gate --

__global__ void gate_kernel(const float* __restrict__ vp,
                            const float* __restrict__ oh,
                            bf16_t* __restrict__ g,
                            int L, int H, int DH, long total) {
  long idx = (long)blockIdx.x * blockDim.x + threadIdx.x;
  long stride = (long)gridDim.x * blockDim.x;
  int DE = H * DH;
  for (; idx < total; idx += stride) {
    int d = (int)(idx % DH);
    long r1 = idx / DH;
    int h = (int)(r1 % H);
    long r2 = r1 / H;
    int l = (int)(r2 % L);
    int b = (int)(r2 / L);
    float pv = vp[((size_t)(b * L + l)) * (size_t)(2 * DE) + DE + (size_t)h * DH + d];
    float ov = oh[(((size_t)(b * H + h)) * L + l) * DH + d];
    float silu = pv / (1.0f + expf(-pv));
    g[((size_t)(b * L + l)) * DE + (size_t)h * DH + d] = (bf16_t)(silu * ov);
  }
}

// -------------------------------------------------------------- output LN --

__global__ void ln_out_kernel(const float* __restrict__ y,
                              const float* __restrict__ w, const float* __restrict__ b,
                              float* __restrict__ out, int D) {
  int row = blockIdx.x;
  int tid = threadIdx.x;
  const float* yr = y + (size_t)row * D;
  float v[4];
  float s = 0.f, ss = 0.f;
#pragma unroll
  for (int j = 0; j < 4; ++j) {
    int d = tid + 256 * j;
    v[j] = yr[d];
    s += v[j]; ss += v[j] * v[j];
  }
  s = wave_rsum(s); ss = wave_rsum(ss);
  __shared__ float red[2][8];
  int wid = tid >> 5, lane = tid & 31;
  if (lane == 0) { red[0][wid] = s; red[1][wid] = ss; }
  __syncthreads();
  s = 0.f; ss = 0.f;
#pragma unroll
  for (int j = 0; j < 8; ++j) { s += red[0][j]; ss += red[1][j]; }
  float mu = s / (float)D;
  float var = ss / (float)D - mu * mu;
  float rs = rsqrtf(var + LN_EPS);
#pragma unroll
  for (int j = 0; j < 4; ++j) {
    int d = tid + 256 * j;
    out[(size_t)row * D + d] = (v[j] - mu) * rs * w[d] + b[d];
  }
}

// ------------------------------------------------------------------ launch --

extern "C" void kernel_launch(void* const* d_in, const int* in_sizes, int n_in,
                              void* d_out, int out_size, void* d_ws, size_t ws_size,
                              hipStream_t stream) {
  (void)in_sizes; (void)n_in; (void)out_size; (void)ws_size;
  const float* x        = (const float*)d_in[0];
  const float* in_ln_w  = (const float*)d_in[1];
  const float* in_ln_b  = (const float*)d_in[2];
  const float* q_ln_w   = (const float*)d_in[3];
  const float* q_ln_b   = (const float*)d_in[4];
  const float* k_ln_w   = (const float*)d_in[5];
  const float* k_ln_b   = (const float*)d_in[6];
  const float* out_ln_w = (const float*)d_in[7];
  const float* out_ln_b = (const float*)d_in[8];
  const float* W_in     = (const float*)d_in[9];
  const float* W_out    = (const float*)d_in[10];
  const float* Wq       = (const float*)d_in[11];
  const float* bq       = (const float*)d_in[12];
  const float* Wk       = (const float*)d_in[13];
  const float* bk       = (const float*)d_in[14];
  const float* smear    = (const float*)d_in[15];
  const float* logsc    = (const float*)d_in[16];

  const int B = 2, L = 2048, DM = 1024, H = 16, DH = 128;
  const int DE = 2048;             // d_exp
  const int M = B * L;             // 4096 rows

  char* ws = (char*)d_ws;
  size_t off = 0;
  auto alloc = [&](size_t bytes) -> void* {
    void* p = ws + off;
    off = (off + bytes + 255) & ~(size_t)255;
    return p;
  };
  bf16_t* Win_t  = (bf16_t*)alloc((size_t)DM * 2 * DE * sizeof(bf16_t));  // [2DE, DM]
  bf16_t* Wq_t   = (bf16_t*)alloc((size_t)DM * DE * sizeof(bf16_t));     // [DE, DM]
  bf16_t* Wk_t   = (bf16_t*)alloc((size_t)DM * DE * sizeof(bf16_t));
  bf16_t* Wout_t = (bf16_t*)alloc((size_t)DE * DM * sizeof(bf16_t));     // [DM, DE]
  bf16_t* a_in   = (bf16_t*)alloc((size_t)M * DM * sizeof(bf16_t));
  bf16_t* a_q    = (bf16_t*)alloc((size_t)M * DM * sizeof(bf16_t));
  bf16_t* a_k    = (bf16_t*)alloc((size_t)M * DM * sizeof(bf16_t));
  float*  vp     = (float*) alloc((size_t)M * 2 * DE * sizeof(float));
  bf16_t* qg     = (bf16_t*)alloc((size_t)M * DE * sizeof(bf16_t));
  bf16_t* kg     = (bf16_t*)alloc((size_t)M * DE * sizeof(bf16_t));
  bf16_t* qhd    = (bf16_t*)alloc((size_t)B * H * L * DH * sizeof(bf16_t));
  bf16_t* khd    = (bf16_t*)alloc((size_t)B * H * L * DH * sizeof(bf16_t));
  bf16_t* vT     = (bf16_t*)alloc((size_t)B * H * DH * L * sizeof(bf16_t));
  float*  ohd    = (float*) alloc((size_t)B * H * L * DH * sizeof(float));
  bf16_t* g      = (bf16_t*)alloc((size_t)M * DE * sizeof(bf16_t));
  float*  y      = (float*) alloc((size_t)M * DM * sizeof(float));

  // weights -> transposed bf16 ([N,K] row-major)
  transpose_w_kernel<<<dim3(2 * DE / 32, DM / 32), 256, 0, stream>>>(W_in,  Win_t,  DM, 2 * DE);
  transpose_w_kernel<<<dim3(DE / 32,     DM / 32), 256, 0, stream>>>(Wq,    Wq_t,   DM, DE);
  transpose_w_kernel<<<dim3(DE / 32,     DM / 32), 256, 0, stream>>>(Wk,    Wk_t,   DM, DE);
  transpose_w_kernel<<<dim3(DM / 32,     DE / 32), 256, 0, stream>>>(W_out, Wout_t, DE, DM);

  // LNs (shared stats, 3 affines)
  ln3_kernel<<<M, 256, 0, stream>>>(x, in_ln_w, in_ln_b, q_ln_w, q_ln_b,
                                    k_ln_w, k_ln_b, a_in, a_q, a_k, DM);

  // vp = ln_in(x) @ W_in  -> f32 [4096, 4096]
  gemm_bf16_kernel<<<(M / 16) * (2 * DE / 64) / 8, 256, 0, stream>>>(
      a_in, Win_t, nullptr, vp, nullptr, M, 2 * DE, DM);
  // q,k GEMMs with bias -> bf16 [4096, 2048]
  gemm_bf16_kernel<<<(M / 16) * (DE / 64) / 8, 256, 0, stream>>>(
      a_q, Wq_t, bq, nullptr, qg, M, DE, DM);
  gemm_bf16_kernel<<<(M / 16) * (DE / 64) / 8, 256, 0, stream>>>(
      a_k, Wk_t, bk, nullptr, kg, M, DE, DM);

  // feature maps + smear + head layout; V head-transpose from vp
  qk_feat_kernel<<<(B * L * H) / 8, 256, 0, stream>>>(
      qg, kg, smear, logsc, qhd, khd, B, L, H, DH);
  transpose_v_kernel<<<dim3(DH / 32, L / 32, B * H), 256, 0, stream>>>(
      vp, vT, L, H, DH, 2 * DE);

  // causal masked attention with row-sum normalization
  attn_kernel<<<(B * H * (L / 16)) / 8, 256, 0, stream>>>(qhd, khd, vT, ohd, L, H);

  // gate
  gate_kernel<<<4096, 256, 0, stream>>>(vp, ohd, g, L, H, DH, (long)M * DE);

  // out GEMM + final LN
  gemm_bf16_kernel<<<(M / 16) * (DM / 64) / 8, 256, 0, stream>>>(
      g, Wout_t, nullptr, y, nullptr, M, DM, DE);
  ln_out_kernel<<<M, 256, 0, stream>>>(y, out_ln_w, out_ln_b, (float*)d_out, DM);
}
